// PostProcessor_13615046329057
// MI455X (gfx1250) — matrix operand order repaired and checked by reading, not compiled, not run on hardware
//
#include <hip/hip_runtime.h>

// ---------------- problem constants (match reference) ----------------
#define NPROP   1024
#define NCLS    32
#define NFG     31            // foreground classes
#define FEATD   4096
#define TOPC    300           // per-class topN after NMS
#define DETS    100           // detections per image
#define NCAND   (NFG * TOPC)  // 9300
#define SCORE_T 0.05f
#define NMS_T   0.5f
#define XCLIP   4.135166556742356f   // log(1000/16)

typedef __attribute__((ext_vector_type(16))) _Float16 v16h;
typedef __attribute__((ext_vector_type(8)))  float    v8f;

// ---------------- wave32 reductions ----------------
__device__ __forceinline__ float waveMax32(float v) {
#pragma unroll
  for (int m = 16; m >= 1; m >>= 1) v = fmaxf(v, __shfl_xor(v, m, 32));
  return v;
}
__device__ __forceinline__ float waveSum32(float v) {
#pragma unroll
  for (int m = 16; m >= 1; m >>= 1) v += __shfl_xor(v, m, 32);
  return v;
}

// ---------------- K0: init candidate arrays ----------------
__global__ void k_init(float* __restrict__ cscore, float* __restrict__ cbox,
                       int* __restrict__ corig, int* __restrict__ clabel) {
  int i = blockIdx.x * blockDim.x + threadIdx.x;
  if (i < NCAND) {
    cscore[i] = -1.0f;
    corig[i]  = 0;
    clabel[i] = 0;
#pragma unroll
    for (int d = 0; d < 6; d++) cbox[i * 6 + d] = 0.0f;
  }
}

// ---------------- K1: softmax + box decode + clip ----------------
// one wave (32 lanes) per proposal, lane == class
__global__ void k_decode(const float* __restrict__ logits,
                         const float* __restrict__ reg,
                         const float* __restrict__ pbox,
                         float* __restrict__ scores,   // [NFG][NPROP]  (-1 if <= thresh)
                         float* __restrict__ boxes) {  // [NFG][NPROP][6]
  int n = blockIdx.x;
  int c = threadIdx.x;  // 0..31
  float lg = logits[n * NCLS + c];
  float mx = waveMax32(lg);
  float e  = __expf(lg - mx);
  float sm = waveSum32(e);
  float prob = e / sm;
  if (c == 0) return;  // background

  const float lim[3] = {95.0f, 319.0f, 319.0f};
  int rb = n * (NCLS * 6) + c * 6;
  int ob = ((c - 1) * NPROP + n) * 6;
#pragma unroll
  for (int d = 0; d < 3; d++) {
    float lo  = pbox[n * 6 + d];
    float hi  = pbox[n * 6 + 3 + d];
    float sz  = hi - lo + 1.0f;
    float ctr = lo + 0.5f * sz;
    float dc  = reg[rb + d] * 0.1f;                     // / 10
    float ds  = fminf(reg[rb + 3 + d] * 0.2f, XCLIP);   // / 5, clip
    float pc  = dc * sz + ctr;
    float ps  = __expf(ds) * sz;
    float mn  = pc - 0.5f * ps;
    float mxv = pc + 0.5f * ps - 1.0f;
    mn  = fminf(fmaxf(mn, 0.0f), lim[d]);
    mxv = fminf(fmaxf(mxv, 0.0f), lim[d]);
    boxes[ob + d]     = mn;
    boxes[ob + 3 + d] = mxv;
  }
  scores[(c - 1) * NPROP + n] = (prob > SCORE_T) ? prob : -1.0f;
}

// ---------------- K2: per-class bitonic sort (desc score, asc idx) ----------------
// sorted boxes written with 8-float (32B) row stride so the NMS kernel can stage
// them with naturally-aligned async B128/B64 transfers.
__global__ void k_sort(const float* __restrict__ scores,
                       const float* __restrict__ boxes,
                       float* __restrict__ sscore, int* __restrict__ sorder,
                       float* __restrict__ sboxes) {   // [NFG*NPROP][8]
  __shared__ float key[NPROP];
  __shared__ int   sid[NPROP];
  int cls = blockIdx.x, tid = threadIdx.x;
  for (int i = tid; i < NPROP; i += 256) { key[i] = scores[cls * NPROP + i]; sid[i] = i; }
  __syncthreads();
  for (int kk = 2; kk <= NPROP; kk <<= 1) {
    for (int jj = kk >> 1; jj > 0; jj >>= 1) {
      for (int i = tid; i < NPROP; i += 256) {
        int ixj = i ^ jj;
        if (ixj > i) {
          float a = key[i], b = key[ixj];
          int ia = sid[i], ib = sid[ixj];
          bool lessBA = (b > a) || (b == a && ib < ia);  // b ranks before a
          bool lessAB = (a > b) || (a == b && ia < ib);
          bool up = ((i & kk) == 0);
          if (up ? lessBA : lessAB) {
            key[i] = b; key[ixj] = a; sid[i] = ib; sid[ixj] = ia;
          }
        }
      }
      __syncthreads();
    }
  }
  for (int i = tid; i < NPROP; i += 256) {
    sscore[cls * NPROP + i] = key[i];
    int src = sid[i];
    sorder[cls * NPROP + i] = src;
#pragma unroll
    for (int d = 0; d < 6; d++)
      sboxes[(size_t)(cls * NPROP + i) * 8 + d] = boxes[(cls * NPROP + src) * 6 + d];
    sboxes[(size_t)(cls * NPROP + i) * 8 + 6] = 0.0f;
    sboxes[(size_t)(cls * NPROP + i) * 8 + 7] = 0.0f;
  }
}

// ---------------- K3: per-class greedy NMS, LDS resident ----------------
// Boxes staged memory->LDS with gfx1250 async-to-LDS ops (ASYNCcnt tracked).
__global__ void k_nms(const float* __restrict__ sscore,
                      const float* __restrict__ sboxes,   // [NFG*NPROP][8]
                      const int* __restrict__ sorder,
                      float* __restrict__ cscore, float* __restrict__ cbox,
                      int* __restrict__ corig, int* __restrict__ clabel) {
  __shared__ float bx[NPROP][8];     // 32 KB, 32B rows
  __shared__ float vol[NPROP];
  __shared__ int   keep[NPROP];
  __shared__ int   scan[NPROP];
  __shared__ int   nvalid_s;
  int cls = blockIdx.x, j = threadIdx.x;

  // async stage: 24 B of box data per lane, direct to LDS (no VGPR round-trip).
  // Low 32 bits of the generic shared address are the LDS byte address (ISA 10.2).
  {
    unsigned ldsAddr = (unsigned)(size_t)(&bx[j][0]);
    unsigned long long ga =
        (unsigned long long)(const void*)(sboxes + (size_t)(cls * NPROP + j) * 8);
    asm volatile("global_load_async_to_lds_b128 %0, %1, off"
                 :: "v"(ldsAddr), "v"(ga) : "memory");
    asm volatile("global_load_async_to_lds_b64 %0, %1, off offset:16"
                 :: "v"(ldsAddr), "v"(ga) : "memory");
  }

  float sj = sscore[cls * NPROP + j];   // overlaps with async transfer
  int kp = (sj > 0.0f) ? 1 : 0;
  keep[j] = kp;
  if (j == 0) nvalid_s = 0;

  asm volatile("s_wait_asynccnt 0x0" ::: "memory");  // this wave's LDS rows ready
  float vz = fmaxf(bx[j][3] - bx[j][0] + 1.0f, 0.0f);
  float vy = fmaxf(bx[j][4] - bx[j][1] + 1.0f, 0.0f);
  float vx = fmaxf(bx[j][5] - bx[j][2] + 1.0f, 0.0f);
  vol[j] = vz * vy * vx;
  __syncthreads();                                   // publish across all 32 waves
  if (kp) atomicAdd(&nvalid_s, 1);
  __syncthreads();
  int nv = nvalid_s;  // sorted desc => valid prefix; block-uniform

  for (int i = 0; i < nv; i++) {
    __syncthreads();
    if (!keep[i]) continue;  // uniform branch (all lanes read same LDS word)
    if (j > i && keep[j]) {
      float ltz = fmaxf(bx[i][0], bx[j][0]);
      float lty = fmaxf(bx[i][1], bx[j][1]);
      float ltx = fmaxf(bx[i][2], bx[j][2]);
      float rbz = fminf(bx[i][3], bx[j][3]);
      float rby = fminf(bx[i][4], bx[j][4]);
      float rbx = fminf(bx[i][5], bx[j][5]);
      float iz = fmaxf(rbz - ltz + 1.0f, 0.0f);
      float iy = fmaxf(rby - lty + 1.0f, 0.0f);
      float ix = fmaxf(rbx - ltx + 1.0f, 0.0f);
      float inter = iz * iy * ix;
      float iou = inter / (vol[i] + vol[j] - inter + 1e-12f);
      if (iou > NMS_T) keep[j] = 0;
    }
  }
  __syncthreads();

  // inclusive Hillis-Steele scan of keep -> rank
  scan[j] = keep[j];
  __syncthreads();
  for (int off = 1; off < NPROP; off <<= 1) {
    int v = scan[j];
    int add = (j >= off) ? scan[j - off] : 0;
    __syncthreads();
    scan[j] = v + add;
    __syncthreads();
  }
  int rank = scan[j] - 1;
  if (keep[j] && rank < TOPC) {
    int o = cls * TOPC + rank;
    cscore[o] = sj;
#pragma unroll
    for (int d = 0; d < 6; d++) cbox[o * 6 + d] = bx[j][d];
    corig[o]  = sorder[cls * NPROP + j];
    clabel[o] = cls + 1;
  }
}

// ---------------- K4: global top-100 by rank counting ----------------
__global__ void k_topk(const float* __restrict__ cscore, int* __restrict__ topIdx) {
  __shared__ float ls[256];
  int i = blockIdx.x * 256 + threadIdx.x;
  float si = (i < NCAND) ? cscore[i] : -1e30f;
  int rank = 0;
  for (int base = 0; base < NCAND; base += 256) {
    int t = base + threadIdx.x;
    ls[threadIdx.x] = (t < NCAND) ? cscore[t] : -1e30f;
    __syncthreads();
    int lim = (NCAND - base) < 256 ? (NCAND - base) : 256;
    for (int jj = 0; jj < lim; jj++) {
      float sjv = ls[jj];
      int jg = base + jj;
      rank += (int)((sjv > si) || (sjv == si && jg < i));
    }
    __syncthreads();
  }
  if (i < NCAND && rank < DETS) topIdx[rank] = i;
}

// ---------------- K5: small outputs + gather indices ----------------
__global__ void k_small(const float* __restrict__ cscore, const float* __restrict__ cbox,
                        const int* __restrict__ corig, const int* __restrict__ clabel,
                        const int* __restrict__ topIdx,
                        int* __restrict__ orig100, float* __restrict__ out) {
  int k = threadIdx.x;
  if (k >= DETS) return;
  int idx = topIdx[k];
  float s = cscore[idx];
  bool val = s > 0.0f;
  float* pboxo = out + (size_t)DETS * FEATD;   // 409600
  float* pscr  = pboxo + DETS * 6;             // 410200
  float* plab  = pscr + DETS;                  // 410300
  float* pval  = plab + DETS;                  // 410400
  pscr[k] = val ? s : 0.0f;
  plab[k] = val ? (float)clabel[idx] : 0.0f;
  pval[k] = val ? 1.0f : 0.0f;
#pragma unroll
  for (int d = 0; d < 6; d++) pboxo[k * 6 + d] = val ? cbox[idx * 6 + d] : 0.0f;
  orig100[k] = val ? corig[idx] : -1;
}

// ---------------- K6: features -> f16 hi/lo, pre-swizzled to WMMA B layout ----------
// B tile (32x16, 16-bit), packed in this code's contiguous k order: lanes 0-15 hold K=0..15 of column n=lane (16 halves),
// lanes 16-31 hold K=16..31 of column n=lane-16.  Flat index == thread id.
__global__ void k_split(const float* __restrict__ feat,
                        _Float16* __restrict__ bhi, _Float16* __restrict__ blo) {
  int t = blockIdx.x * 256 + threadIdx.x;       // 0 .. 1024*4096-1
  int e    = t & 15;
  int lane = (t >> 4) & 31;
  int nt   = (t >> 9) & 255;
  int kt   = t >> 17;
  int k = kt * 32 + ((lane < 16) ? e : 16 + e); // proposal (K dim)
  int n = nt * 16 + (lane & 15);                // feature column
  float f = feat[(size_t)k * FEATD + n];
  _Float16 hi = (_Float16)f;
  _Float16 lo = (_Float16)(f - (float)hi);
  bhi[t] = hi;
  blo[t] = lo;
}

// ---------------- K7: one-hot gather as split-precision WMMA GEMM ----------------
// D[100x4096] = S[100x1024] x F[1024x4096];  S one-hot at orig100[m] (exact in f16),
// F = hi + lo, accumulated in f32:  D = S*hi + S*lo.
__global__ void k_gather_wmma(const _Float16* __restrict__ bhi,
                              const _Float16* __restrict__ blo,
                              const int* __restrict__ orig100,
                              float* __restrict__ outFeat) {
  int mt = blockIdx.x;          // 0..6  (rows, 112 padded)
  int nt = blockIdx.y;          // 0..255 (feature cols / 16)
  int lane = threadIdx.x;       // single wave32
  int m  = mt * 16 + (lane & 15);
  int om = (m < DETS) ? orig100[m] : -1;   // -1 => zero row

  v8f c = {};
  for (int kt = 0; kt < 32; kt++) {
    // wave-uniform skip: no lane's selected proposal falls in this K tile
    bool match = (om >= 0) && ((om >> 5) == kt);
    if (__ballot(match) == 0ull) continue;

    size_t bo = ((size_t)(kt * 256 + nt) * 32 + lane) * 16;
    if (kt + 1 < 32)
      __builtin_prefetch(bhi + bo + (size_t)256 * 32 * 16, 0, 0);  // global_prefetch_b8

    // Build one-hot A slice in contiguous k order (same order as the B packing):
    // lane<16: K=e ; lane>=16: K=16+e
    v16h a;
#pragma unroll
    for (int e = 0; e < 16; e++) {
      int K = e + ((lane >= 16) ? 16 : 0);
      a[e] = (om == kt * 32 + K) ? (_Float16)1.0f : (_Float16)0.0f;
    }
    v16h bh = *(const v16h*)(bhi + bo);
    v16h bl = *(const v16h*)(blo + bo);
    c = __builtin_amdgcn_wmma_f32_16x16x32_f16(false, a, false, bh, (short)0, c, false, false);
    c = __builtin_amdgcn_wmma_f32_16x16x32_f16(false, a, false, bl, (short)0, c, false, false);
  }

  int col   = nt * 16 + (lane & 15);
  int rbase = mt * 16 + ((lane >= 16) ? 8 : 0);
#pragma unroll
  for (int v = 0; v < 8; v++) {
    int r = rbase + v;
    if (r < DETS) outFeat[(size_t)r * FEATD + col] = c[v];
  }
}

// ---------------- host launch ----------------
extern "C" void kernel_launch(void* const* d_in, const int* in_sizes, int n_in,
                              void* d_out, int out_size, void* d_ws, size_t ws_size,
                              hipStream_t stream) {
  (void)in_sizes; (void)n_in; (void)out_size; (void)ws_size;
  const float* features = (const float*)d_in[0];
  const float* logits   = (const float*)d_in[1];
  const float* reg      = (const float*)d_in[2];
  const float* pbox     = (const float*)d_in[3];
  float* out = (float*)d_out;

  char* ws = (char*)d_ws;
  size_t off = 0;
  auto wsAlloc = [&](size_t bytes) -> void* {
    void* p = ws + off;
    off = (off + bytes + 255) & ~(size_t)255;
    return p;
  };
  float* scores  = (float*)wsAlloc((size_t)NFG * NPROP * 4);
  float* boxes   = (float*)wsAlloc((size_t)NFG * NPROP * 24);
  float* sscore  = (float*)wsAlloc((size_t)NFG * NPROP * 4);
  int*   sorder  = (int*)  wsAlloc((size_t)NFG * NPROP * 4);
  float* sboxes  = (float*)wsAlloc((size_t)NFG * NPROP * 32);  // padded 8-float rows
  float* cscore  = (float*)wsAlloc((size_t)NCAND * 4);
  float* cbox    = (float*)wsAlloc((size_t)NCAND * 24);
  int*   corig   = (int*)  wsAlloc((size_t)NCAND * 4);
  int*   clabel  = (int*)  wsAlloc((size_t)NCAND * 4);
  int*   topIdx  = (int*)  wsAlloc(DETS * 4);
  int*   orig100 = (int*)  wsAlloc(DETS * 4);
  _Float16* bhi  = (_Float16*)wsAlloc((size_t)NPROP * FEATD * 2);
  _Float16* blo  = (_Float16*)wsAlloc((size_t)NPROP * FEATD * 2);

  k_init  <<<(NCAND + 255) / 256, 256, 0, stream>>>(cscore, cbox, corig, clabel);
  k_split <<<(NPROP * FEATD) / 256, 256, 0, stream>>>(features, bhi, blo);
  k_decode<<<NPROP, 32, 0, stream>>>(logits, reg, pbox, scores, boxes);
  k_sort  <<<NFG, 256, 0, stream>>>(scores, boxes, sscore, sorder, sboxes);
  k_nms   <<<NFG, NPROP, 0, stream>>>(sscore, sboxes, sorder, cscore, cbox, corig, clabel);
  k_topk  <<<(NCAND + 255) / 256, 256, 0, stream>>>(cscore, topIdx);
  k_small <<<1, 128, 0, stream>>>(cscore, cbox, corig, clabel, topIdx, orig100, out);
  k_gather_wmma<<<dim3(7, 256), 32, 0, stream>>>(bhi, blo, orig100, out);
}
